// WaveGateAttention_103079215211
// MI455X (gfx1250) — matrix operand order repaired and checked by reading, not compiled, not run on hardware
//
#include <hip/hip_runtime.h>
#include <hip/hip_bf16.h>
#include <stdint.h>

// ---------------------------------------------------------------------------
// WaveGateAttention forward for MI455X (gfx1250, wave32, WMMA)
//
// Sizes: V=100277, C=384, K=3, L=4, T=256, B=8  -> M = B*T = 2048 tokens
// All matmuls (conv-as-GEMM, gate, out, head) run through one bf16 WMMA GEMM
// kernel (v_wmma_f32_16x16x32_bf16, f32 accumulate), double-buffered LDS with
// async global->LDS staging (GLOBAL_LOAD_ASYNC_TO_LDS_B128 / ASYNCcnt).
// ---------------------------------------------------------------------------

#define BDIM   8
#define TDIM   256
#define CDIM   384
#define C3DIM  1152
#define KCONV  3
#define LDIM   4
#define VDIM   100277
#define VPAD   100352          // ceil(V/128)*128
#define MDIM   2048            // B*T

typedef __bf16 bf16_t;
typedef __attribute__((ext_vector_type(16))) __bf16 v16bf;
typedef __attribute__((ext_vector_type(8)))  __bf16 v8bf;
typedef __attribute__((ext_vector_type(8)))  float  v8f;
typedef __attribute__((ext_vector_type(4)))  int    v4i;

// Pointer types the async-to-LDS builtin expects (from hipcc diagnostics:
// param 1 is "int __vector(4) __device__ *", i.e. v4i addrspace(1)*).
typedef __attribute__((address_space(1))) v4i* gptr_v4i;
typedef __attribute__((address_space(3))) v4i* lptr_v4i;

#if defined(__has_builtin)
#if __has_builtin(__builtin_amdgcn_global_load_async_to_lds_b128)
#define HAVE_ASYNC_LDS 1
#endif
#endif

__device__ __forceinline__ void wait_async0() {
#if defined(__has_builtin) && __has_builtin(__builtin_amdgcn_s_wait_asynccnt)
    __builtin_amdgcn_s_wait_asynccnt(0);
#else
    asm volatile("s_wait_asynccnt 0" ::: "memory");
#endif
}

// ---------------------------------------------------------------------------
// Generic bf16 GEMM:  Out[m,n] = sum_k A[m,k]*W[n,k]  (+bias[n]) (+resid[m,n])
// A: [M x K] bf16 row-major (K multiple of 32, rows 16B aligned)
// W: [Npad x K] bf16 row-major, Npad multiple of 128 (pad rows zero-filled)
// Block: 256 threads = 8 waves. Block tile 128(M) x 128(N).
// Wave (w%4, w/4) computes a 32x64 sub-tile = 2x4 WMMA 16x16 accumulators.
// Double-buffered 32-wide K panels in LDS (32 KB).
// ---------------------------------------------------------------------------
#define TILE_M 128
#define TILE_N 128
#define KSTEP  32

__device__ __forceinline__ v16bf cat8(v8bf a, v8bf b) {
    v16bf r;
#pragma unroll
    for (int i = 0; i < 8; ++i) { r[i] = a[i]; r[8 + i] = b[i]; }
    return r;
}

__global__ __launch_bounds__(256)
void k_gemm_bf16(const bf16_t* __restrict__ A, const bf16_t* __restrict__ W,
                 const float* __restrict__ bias, const float* __restrict__ resid,
                 float* __restrict__ Out, int M, int N, int K)
{
    __shared__ __align__(16) bf16_t As[2][TILE_M * KSTEP];   // 2 x 8 KB
    __shared__ __align__(16) bf16_t Ws[2][TILE_N * KSTEP];   // 2 x 8 KB

    const int tid  = threadIdx.x;
    const int wave = tid >> 5;       // 0..7
    const int lane = tid & 31;
    const int wm   = wave & 3;       // wave M slot: offset wm*32
    const int wn   = wave >> 2;      // wave N slot: offset wn*64
    const int lr   = lane & 15;      // lane row/col within 16
    const int hi   = lane >> 4;      // lane half (ISA A/B K-split)

    const int blockM = blockIdx.y * TILE_M;
    const int blockN = blockIdx.x * TILE_N;

    // Each thread owns one 64-byte row-slice of either the A or the W panel.
    const int r = tid & 127;
    const bf16_t* gsrc = (tid < 128)
        ? (A + (size_t)(blockM + r) * K)
        : (W + (size_t)(blockN + r) * K);
    bf16_t* lrow0 = ((tid < 128) ? As[0] : Ws[0]) + r * KSTEP;
    bf16_t* lrow1 = ((tid < 128) ? As[1] : Ws[1]) + r * KSTEP;

    v8f acc[2][4];
#pragma unroll
    for (int mt = 0; mt < 2; ++mt)
#pragma unroll
        for (int nt = 0; nt < 4; ++nt)
            acc[mt][nt] = (v8f){0.f, 0.f, 0.f, 0.f, 0.f, 0.f, 0.f, 0.f};

#if defined(HAVE_ASYNC_LDS)
    // ---- async global->LDS pipeline (ASYNCcnt-tracked, no VGPR roundtrip) ----
    {   // prologue: stage buffer 0
        const bf16_t* src = gsrc;
#pragma unroll
        for (int j = 0; j < 4; ++j)
            __builtin_amdgcn_global_load_async_to_lds_b128(
                (gptr_v4i)(src + j * 8), (lptr_v4i)(lrow0 + j * 8), 0, 0);
    }
    int cur = 0;
    for (int k0 = 0; k0 < K; k0 += KSTEP) {
        wait_async0();           // own async stores to LDS done
        __syncthreads();         // => every wave's stores done: panel ready
        if (k0 + KSTEP < K) {    // prefetch next panel into the other buffer
            const bf16_t* src = gsrc + k0 + KSTEP;
            bf16_t* dst = cur ? lrow0 : lrow1;
#pragma unroll
            for (int j = 0; j < 4; ++j)
                __builtin_amdgcn_global_load_async_to_lds_b128(
                    (gptr_v4i)(src + j * 8), (lptr_v4i)(dst + j * 8), 0, 0);
        }
        const bf16_t* Ab = As[cur];
        const bf16_t* Wb = Ws[cur];
#else
    // ---- fallback: register-staged double buffer ----
    {   // prologue: stage buffer 0 synchronously
        const v8bf* s = (const v8bf*)gsrc;
        v8bf*       d = (v8bf*)lrow0;
        d[0] = s[0]; d[1] = s[1]; d[2] = s[2]; d[3] = s[3];
    }
    __syncthreads();
    int cur = 0;
    for (int k0 = 0; k0 < K; k0 += KSTEP) {
        v8bf stage[4];
        const bool has_next = (k0 + KSTEP < K);
        if (has_next) {          // issue global loads early; consume after wmma
            const v8bf* s = (const v8bf*)(gsrc + k0 + KSTEP);
            stage[0] = s[0]; stage[1] = s[1]; stage[2] = s[2]; stage[3] = s[3];
        }
        const bf16_t* Ab = As[cur];
        const bf16_t* Wb = Ws[cur];
#endif

        // A fragment: 16x32 bf16. Lane half 0: K 0-7 & 16-23; half 1: 8-15 & 24-31.
        v16bf afr[2];
#pragma unroll
        for (int mt = 0; mt < 2; ++mt) {
            const int row = wm * 32 + mt * 16 + lr;
            const int ka  = hi ? 8 : 0;
            v8bf lo = *(const v8bf*)(Ab + row * KSTEP + ka);
            v8bf h2 = *(const v8bf*)(Ab + row * KSTEP + ka + 16);
            afr[mt] = cat8(lo, h2);
        }
        // B fragment: 32x16 (KxN). Lane half 0: K 0-7 & 16-23; half 1: 8-15 & 24-31.
        v16bf bfr[4];
#pragma unroll
        for (int nt = 0; nt < 4; ++nt) {
            const int row = wn * 64 + nt * 16 + lr;   // N index within tile
            const int kb  = hi ? 8 : 0;
            v8bf lo = *(const v8bf*)(Wb + row * KSTEP + kb);
            v8bf h2 = *(const v8bf*)(Wb + row * KSTEP + kb + 16);
            bfr[nt] = cat8(lo, h2);
        }

#pragma unroll
        for (int mt = 0; mt < 2; ++mt)
#pragma unroll
            for (int nt = 0; nt < 4; ++nt)
                acc[mt][nt] = __builtin_amdgcn_wmma_f32_16x16x32_bf16(
                    false, afr[mt], false, bfr[nt],
                    (short)0, acc[mt][nt], false, false);

#if defined(HAVE_ASYNC_LDS)
        cur ^= 1;
    }
#else
        if (has_next) {          // store staged panel into the other buffer
            v8bf* d = (v8bf*)(cur ? lrow0 : lrow1);
            d[0] = stage[0]; d[1] = stage[1]; d[2] = stage[2]; d[3] = stage[3];
        }
        __syncthreads();
        cur ^= 1;
    }
#endif

    // Epilogue. D layout: VGPR j, lane -> row = (lane/16)*8 + j, col = lane%16.
    // Logits (821 MB f32) are write-once: use non-temporal stores so L2 keeps
    // the streamed bf16 weight panels instead.
#pragma unroll
    for (int nt = 0; nt < 4; ++nt) {
        const int col = blockN + wn * 64 + nt * 16 + lr;
        if (col < N) {
            const float badd = bias ? bias[col] : 0.f;
#pragma unroll
            for (int mt = 0; mt < 2; ++mt) {
                const int rbase = blockM + wm * 32 + mt * 16 + hi * 8;
#pragma unroll
                for (int j = 0; j < 8; ++j) {
                    const size_t o = (size_t)(rbase + j) * N + col;
                    float v = acc[mt][nt][j] + badd;
                    if (resid) v += resid[o];
                    __builtin_nontemporal_store(v, &Out[o]);
                }
            }
        }
    }
}

// ---------------------------------------------------------------------------
// Elementwise / helper kernels
// ---------------------------------------------------------------------------

__global__ void k_embed(const int* __restrict__ idx, const float* __restrict__ emb,
                        float* __restrict__ x)
{
    int i = blockIdx.x * blockDim.x + threadIdx.x;
    if (i >= MDIM * CDIM) return;
    int m = i / CDIM, c = i % CDIM;
    x[i] = emb[(size_t)idx[m] * CDIM + c];
}

// LayerNorm over C=384, one row per block of 128 threads; write bf16.
__global__ __launch_bounds__(128)
void k_ln(const float* __restrict__ x, const float* __restrict__ g,
          const float* __restrict__ b, bf16_t* __restrict__ out)
{
    const int m   = blockIdx.x;
    const int tid = threadIdx.x;
    const float* row = x + (size_t)m * CDIM;
    float v0 = row[tid], v1 = row[tid + 128], v2 = row[tid + 256];
    float s  = v0 + v1 + v2;
    float ss = v0 * v0 + v1 * v1 + v2 * v2;
#pragma unroll
    for (int off = 16; off; off >>= 1) {
        s  += __shfl_xor(s,  off, 32);
        ss += __shfl_xor(ss, off, 32);
    }
    __shared__ float red[8];
    const int wave = tid >> 5;
    if ((tid & 31) == 0) { red[wave] = s; red[wave + 4] = ss; }
    __syncthreads();
    float ts = red[0] + red[1] + red[2] + red[3];
    float tq = red[4] + red[5] + red[6] + red[7];
    const float mean = ts * (1.f / CDIM);
    const float var  = tq * (1.f / CDIM) - mean * mean;
    const float rstd = rsqrtf(var + 1e-5f);
    bf16_t* orow = out + (size_t)m * CDIM;
    orow[tid]       = (bf16_t)((v0 - mean) * rstd * g[tid]       + b[tid]);
    orow[tid + 128] = (bf16_t)((v1 - mean) * rstd * g[tid + 128] + b[tid + 128]);
    orow[tid + 256] = (bf16_t)((v2 - mean) * rstd * g[tid + 256] + b[tid + 256]);
}

// Build conv-as-GEMM A: Ac[m, k*C+ci] = xn[b, t+k-2, ci] (0 if t+k-2 < 0).
__global__ void k_conv_a(const bf16_t* __restrict__ xn, bf16_t* __restrict__ Ac)
{
    int i = blockIdx.x * blockDim.x + threadIdx.x;
    if (i >= MDIM * C3DIM) return;
    int m = i / C3DIM, j = i % C3DIM;
    int k = j / CDIM, ci = j % CDIM;
    int t = m % TDIM, tt = t + k - (KCONV - 1);
    Ac[i] = (tt >= 0) ? xn[(size_t)(m + k - (KCONV - 1)) * CDIM + ci] : (bf16_t)0.f;
}

// Repack conv_w[l] (C,C,K) -> Wc[co, k*C+ci], bf16.
__global__ void k_conv_w(const float* __restrict__ cw, bf16_t* __restrict__ Wb)
{
    int i = blockIdx.x * blockDim.x + threadIdx.x;
    if (i >= CDIM * C3DIM) return;
    int co = i / C3DIM, j = i % C3DIM;
    int k = j / CDIM, ci = j % CDIM;
    Wb[i] = (bf16_t)cw[((size_t)co * CDIM + ci) * KCONV + k];
}

// f32 -> bf16 weight convert with zero row padding to Npad rows.
__global__ void k_wcvt(const float* __restrict__ W, bf16_t* __restrict__ Wb,
                       int Nreal, int Npad, int K)
{
    int i = blockIdx.x * blockDim.x + threadIdx.x;
    if (i >= Npad * K) return;
    int n = i / K;
    Wb[i] = (n < Nreal) ? (bf16_t)__builtin_nontemporal_load(&W[i]) : (bf16_t)0.f;
}

// Gate nonlinearity: f = sigmoid(g0), ic = sigmoid(g1)*tanh(g2).
__global__ void k_gates(const float* __restrict__ g, float* __restrict__ fs,
                        float* __restrict__ ic)
{
    int i = blockIdx.x * blockDim.x + threadIdx.x;
    if (i >= MDIM * CDIM) return;
    int m = i / CDIM, c = i % CDIM;
    const float* row = g + (size_t)m * C3DIM;
    float fv = row[c], iv = row[CDIM + c], cv = row[2 * CDIM + c];
    fs[i] = 1.f / (1.f + __expf(-fv));
    ic[i] = (1.f / (1.f + __expf(-iv))) * tanhf(cv);
}

// Linear recurrence over T: s_t = f_t * s_{t-1} + ic_t. 3072 independent lanes.
__global__ void k_scan(const float* __restrict__ fs, const float* __restrict__ ic,
                       bf16_t* __restrict__ st)
{
    int i = blockIdx.x * blockDim.x + threadIdx.x;
    if (i >= BDIM * CDIM) return;
    int b = i / CDIM, c = i % CDIM;
    float s = 0.f;
    for (int t = 0; t < TDIM; ++t) {
        size_t o = (size_t)(b * TDIM + t) * CDIM + c;
        s = fs[o] * s + ic[o];
        st[o] = (bf16_t)s;
    }
}

// x += alpha * x * exp(-x^2/2)
__global__ void k_exp(float* __restrict__ x, const float* __restrict__ alpha, int l)
{
    int i = blockIdx.x * blockDim.x + threadIdx.x;
    if (i >= MDIM * CDIM) return;
    float v = x[i], a = alpha[l];
    x[i] = v + a * v * __expf(-v * v * 0.5f);
}

__global__ void k_xcvt(const float* __restrict__ x, bf16_t* __restrict__ xb)
{
    int i = blockIdx.x * blockDim.x + threadIdx.x;
    if (i >= MDIM * CDIM) return;
    xb[i] = (bf16_t)x[i];
}

// ---------------------------------------------------------------------------
// Launch
// ---------------------------------------------------------------------------
static inline int ceil_div(int a, int b) { return (a + b - 1) / b; }

extern "C" void kernel_launch(void* const* d_in, const int* in_sizes, int n_in,
                              void* d_out, int out_size, void* d_ws, size_t ws_size,
                              hipStream_t stream)
{
    const int*   idx    = (const int*)  d_in[0];
    const float* emb    = (const float*)d_in[1];
    const float* conv_w = (const float*)d_in[2];
    const float* conv_b = (const float*)d_in[3];
    const float* gate_w = (const float*)d_in[4];
    const float* out_w  = (const float*)d_in[5];
    const float* alpha  = (const float*)d_in[6];
    const float* n1_g   = (const float*)d_in[7];
    const float* n1_b   = (const float*)d_in[8];
    const float* n2_g   = (const float*)d_in[9];
    const float* n2_b   = (const float*)d_in[10];
    const float* head_w = (const float*)d_in[11];
    const float* head_b = (const float*)d_in[12];
    float* logits = (float*)d_out;

    // Workspace carve-up (256B aligned slices).
    char* p = (char*)d_ws;
    auto carve = [&](size_t bytes) -> void* {
        void* r = (void*)p;
        p += (bytes + 255) & ~(size_t)255;
        return r;
    };
    float*  x    = (float*) carve((size_t)MDIM * CDIM * 4);
    bf16_t* xnb  = (bf16_t*)carve((size_t)MDIM * CDIM * 2);
    bf16_t* Ac   = (bf16_t*)carve((size_t)MDIM * C3DIM * 2);
    bf16_t* Wb   = (bf16_t*)carve((size_t)VPAD * CDIM * 2);   // shared W buffer
    float*  g    = (float*) carve((size_t)MDIM * C3DIM * 4);
    float*  fs   = (float*) carve((size_t)MDIM * CDIM * 4);
    float*  icb  = (float*) carve((size_t)MDIM * CDIM * 4);
    bf16_t* stb  = (bf16_t*)carve((size_t)MDIM * CDIM * 2);

    const int TPB = 256;
    const int nMC  = MDIM * CDIM;     // 786432
    const int nM3C = MDIM * C3DIM;    // 2359296
    const int nWC  = CDIM * C3DIM;    // 442368

    // Embedding gather.
    k_embed<<<ceil_div(nMC, TPB), TPB, 0, stream>>>(idx, emb, x);

    for (int l = 0; l < LDIM; ++l) {
        // --- causal conv block (as one WMMA GEMM over gathered taps) ---
        k_ln<<<MDIM, 128, 0, stream>>>(x, n1_g + l * CDIM, n1_b + l * CDIM, xnb);
        k_conv_a<<<ceil_div(nM3C, TPB), TPB, 0, stream>>>(xnb, Ac);
        k_conv_w<<<ceil_div(nWC, TPB), TPB, 0, stream>>>(
            conv_w + (size_t)l * CDIM * CDIM * KCONV, Wb);
        {   // x = x + conv(xn) + conv_b
            dim3 grid(CDIM / TILE_N, MDIM / TILE_M);
            k_gemm_bf16<<<grid, 256, 0, stream>>>(Ac, Wb, conv_b + l * CDIM,
                                                  x, x, MDIM, CDIM, C3DIM);
        }

        // --- gated linear recurrence ---
        k_ln<<<MDIM, 128, 0, stream>>>(x, n2_g + l * CDIM, n2_b + l * CDIM, xnb);
        k_wcvt<<<ceil_div(C3DIM * CDIM, TPB), TPB, 0, stream>>>(
            gate_w + (size_t)l * C3DIM * CDIM, Wb, C3DIM, C3DIM, CDIM);
        {   // g = xn @ gate_w^T
            dim3 grid(C3DIM / TILE_N, MDIM / TILE_M);
            k_gemm_bf16<<<grid, 256, 0, stream>>>(xnb, Wb, nullptr, nullptr,
                                                  g, MDIM, C3DIM, CDIM);
        }
        k_gates<<<ceil_div(nMC, TPB), TPB, 0, stream>>>(g, fs, icb);
        k_scan<<<ceil_div(BDIM * CDIM, TPB), TPB, 0, stream>>>(fs, icb, stb);
        k_wcvt<<<ceil_div(CDIM * CDIM, TPB), TPB, 0, stream>>>(
            out_w + (size_t)l * CDIM * CDIM, Wb, CDIM, CDIM, CDIM);
        {   // x = x + states @ out_w^T
            dim3 grid(CDIM / TILE_N, MDIM / TILE_M);
            k_gemm_bf16<<<grid, 256, 0, stream>>>(stb, Wb, nullptr,
                                                  x, x, MDIM, CDIM, CDIM);
        }

        // --- exponential transform ---
        k_exp<<<ceil_div(nMC, TPB), TPB, 0, stream>>>(x, alpha, l);
    }

    // --- head: logits = x @ head_w^T + head_b (dominant GEMM) ---
    k_xcvt<<<ceil_div(nMC, TPB), TPB, 0, stream>>>(x, xnb);
    k_wcvt<<<ceil_div(VPAD * CDIM, TPB), TPB, 0, stream>>>(
        head_w, Wb, VDIM, VPAD, CDIM);
    {
        dim3 grid(VPAD / TILE_N, MDIM / TILE_M);
        k_gemm_bf16<<<grid, 256, 0, stream>>>(xnb, Wb, head_b, nullptr,
                                              logits, MDIM, VDIM, CDIM);
    }
}
